// comb_encoder_33225867002577
// MI455X (gfx1250) — matrix operand order repaired and checked by reading, not compiled, not run on hardware
//
#include <hip/hip_runtime.h>
#include <math.h>

// ---------------------------------------------------------------------------
// Bidirectional LSTM (B=8,S=1024,D=H=512) + masked neighbor combine, gfx1250.
//
// Strategy (latency-bound recurrence, M=8 fits in one 16x16 WMMA M-tile):
//  * 32 workgroups total: 16 per direction, each owns 32 hidden units
//    (4 gates x 32 cols = 8 N-tiles of 16), 8 waves x 1 N-tile each.
//  * W_hh B-fragments preloaded into VGPRs (16 k-steps x v16bf = 128 VGPRs).
//  * W_ih B-fragments packed into LDS (128KB) in exact WMMA lane layout.
//  * x pre-converted to bf16 with rows 8..15 zero-padded (A-tile pad rows).
//  * h (8x512 bf16) double-buffered in global; per-step release/acquire via
//    a monotonic atomic counter barrier across the direction's 16 WGs.
//  * c state lives in LDS (fp32); gates computed in fp32.
//  * WMMA: v_wmma_f32_16x16x32_bf16, fp32 accum.
// ---------------------------------------------------------------------------

typedef __bf16 bf16_t;
typedef __attribute__((ext_vector_type(16))) __bf16 v16bf;
typedef __attribute__((ext_vector_type(8)))  __bf16 v8bf;
typedef __attribute__((ext_vector_type(8)))  float  v8f;

#define S_LEN 1024
#define B_SZ  8
#define D_SZ  512
#define H_SZ  512
#define NWG_PER_DIR 16
#define THREADS_LSTM 256   // 8 wave32

// workspace offsets (bytes); total ~50.4 MB
#define X_BF_OFF   0ull          // 1024 x 16 x 512 bf16   = 16 MB
#define HBUF_OFF   16777216ull   // 2 dir x 2 buf x 16x512 bf16 = 64 KB
#define HSF_OFF    16842752ull   // 1024 x 8 x 512 f32     = 16 MB
#define HSB_OFF    33619968ull   // 1024 x 8 x 512 f32     = 16 MB
#define BAR_OFF    50397184ull   // 2 x 64 uints

// d_out layout (floats): out(8x1022x1024) ++ h_n(2x8x512) ++ c_n(2x8x512)
#define OUT_HN 8372224
#define OUT_CN 8380416

// LDS partition (dynamic): W_ih tiles 131072B | zscr 4096B | cstate 1024B
#define LDS_WIH   131072
#define LDS_ZSCR  4096
#define LDS_CST   1024

__device__ __forceinline__ float sigmoidf_(float x) {
  return 1.0f / (1.0f + __expf(-x));
}

// A-fragment (16x32 bf16) for wave32 WMMA, from row-major [16][512] bf16:
//   lane<16 : row=lane,    k = kbase + {0..15}
//   lane>=16: row=lane-16, k = kbase + {16..31}
__device__ __forceinline__ v16bf load_A_frag(const bf16_t* __restrict__ base,
                                             int lane, int kbase) {
  int m  = lane & 15;
  int hi = lane >> 4;
  const bf16_t* p = base + m * 512 + kbase + hi * 16;
  v8bf lo = *(const v8bf*)(p);
  v8bf hv = *(const v8bf*)(p + 8);
  v16bf r;
#pragma unroll
  for (int e = 0; e < 8; ++e) { r[e] = lo[e]; r[8 + e] = hv[e]; }
  return r;
}

// ---------------------------------------------------------------------------
__global__ void build_xbf(const float* __restrict__ x, bf16_t* __restrict__ xbf) {
  size_t i = (size_t)blockIdx.x * blockDim.x + threadIdx.x; // over 1024*16*512
  int t = (int)(i >> 13);
  int m = (int)((i >> 9) & 15);
  int d = (int)(i & 511);
  float v = (m < B_SZ) ? x[((size_t)m * S_LEN + t) * D_SZ + d] : 0.0f;
  xbf[i] = (bf16_t)v;
}

__global__ void init_small(bf16_t* __restrict__ hbuf, unsigned* __restrict__ bar) {
  for (int i = threadIdx.x; i < 2 * 2 * 16 * 512; i += blockDim.x)
    hbuf[i] = (bf16_t)0.0f;
  if (threadIdx.x < 128) bar[threadIdx.x] = 0u;
}

// ---------------------------------------------------------------------------
__global__ void __launch_bounds__(THREADS_LSTM)
lstm_kernel(const bf16_t* __restrict__ xbf,
            bf16_t* __restrict__ hbuf,
            float* __restrict__ hsf, float* __restrict__ hsb,
            const float* __restrict__ wih_f, const float* __restrict__ whh_f,
            const float* __restrict__ bias_f,
            const float* __restrict__ wih_b, const float* __restrict__ whh_b,
            const float* __restrict__ bias_b,
            unsigned* __restrict__ bar, float* __restrict__ out)
{
  extern __shared__ char smem[];
  bf16_t* wih_lds = (bf16_t*)smem;                          // packed B tiles
  float*  zscr    = (float*)(smem + LDS_WIH);               // [4][32][8]
  float*  cstate  = (float*)(smem + LDS_WIH + LDS_ZSCR);    // [32][8]

  const int dir  = blockIdx.x >> 4;   // 0 = fwd, 1 = bwd
  const int wg   = blockIdx.x & 15;
  const int j0   = wg * 32;           // hidden-unit base for this WG
  const int tid  = threadIdx.x;
  const int wave = tid >> 5, lane = tid & 31;
  const int gate = wave >> 1;
  const int jjb  = (wave & 1) << 4;

  const float* wih = dir ? wih_b : wih_f;
  const float* whh = dir ? whh_b : whh_f;
  const float* bia = dir ? bias_b : bias_f;
  float* hs = dir ? hsb : hsf;
  unsigned* cntp = bar + dir * 64;
  unsigned* genp = bar + dir * 64 + 32;

  // ---- pack this WG's W_ih slice into LDS in WMMA B-fragment order -------
  // row r = (wave, kstep, lane); each row = 16 contiguous k of one column.
  for (int r = tid; r < 4096; r += THREADS_LSTM) {
    int wv = r >> 9, ks = (r >> 5) & 15, ln = r & 31;
    int col = (wv >> 1) * 512 + j0 + ((wv & 1) << 4) + (ln & 15);
    int k   = ks * 32 + ((ln >> 4) << 4);
    const float* src = wih + (size_t)col * 512 + k;
    bf16_t* dst = wih_lds + (size_t)r * 16;
#pragma unroll
    for (int e = 0; e < 16; ++e) dst[e] = (bf16_t)src[e];
  }
  cstate[tid] = 0.0f;

  // ---- preload W_hh B-fragments into VGPRs (constant across all steps) ---
  const int colw = gate * 512 + j0 + jjb + (lane & 15);
  v16bf whh_reg[16];
#pragma unroll
  for (int ks = 0; ks < 16; ++ks) {
    int k = ks * 32 + ((lane >> 4) << 4);
    const float* src = whh + (size_t)colw * 512 + k;
    v16bf v;
#pragma unroll
    for (int e = 0; e < 16; ++e) v[e] = (bf16_t)src[e];
    whh_reg[ks] = v;
  }

  // per-thread gate biases for the elementwise phase
  const int jj_e = tid >> 3, b_e = tid & 7;
  const float bi0 = bia[0 * 512 + j0 + jj_e];
  const float bi1 = bia[1 * 512 + j0 + jj_e];
  const float bi2 = bia[2 * 512 + j0 + jj_e];
  const float bi3 = bia[3 * 512 + j0 + jj_e];

  __syncthreads();

  for (int s = 0; s < S_LEN; ++s) {
    const int t = dir ? (S_LEN - 1 - s) : s;
    const bf16_t* xA = xbf  + (size_t)t * (16 * 512);
    const bf16_t* hA = hbuf + (size_t)(dir * 2 + (s & 1)) * (16 * 512);
    bf16_t*       hN = hbuf + (size_t)(dir * 2 + ((s + 1) & 1)) * (16 * 512);

    v8f acc = {0.f, 0.f, 0.f, 0.f, 0.f, 0.f, 0.f, 0.f};

    // z = x_t @ W_ih^T  (B tiles from LDS, conflict-free 32B reads)
#pragma unroll
    for (int ks = 0; ks < 16; ++ks) {
      v16bf a = load_A_frag(xA, lane, ks * 32);
      v16bf b = *(const v16bf*)(wih_lds +
                 (size_t)(((wave * 16 + ks) * 32 + lane) * 16));
      acc = __builtin_amdgcn_wmma_f32_16x16x32_bf16(
              false, a, false, b, (short)0, acc, false, false);
    }
    //   += h_{t-1} @ W_hh^T  (B tiles from registers)
#pragma unroll
    for (int ks = 0; ks < 16; ++ks) {
      v16bf a = load_A_frag(hA, lane, ks * 32);
      acc = __builtin_amdgcn_wmma_f32_16x16x32_bf16(
              false, a, false, whh_reg[ks], (short)0, acc, false, false);
    }

    // scatter C fragment (rows 0..7 valid) into LDS: zscr[gate][jj][b]
    if (lane < 16) {
      float* zc = zscr + ((gate * 32) + (jjb + lane)) * 8;
#pragma unroll
      for (int r = 0; r < 8; ++r) zc[r] = acc[r];
    }
    __syncthreads();

    // elementwise LSTM cell: one thread per (hidden unit jj, batch b)
    {
      float zi = zscr[(0 * 32 + jj_e) * 8 + b_e] + bi0;
      float zf = zscr[(1 * 32 + jj_e) * 8 + b_e] + bi1;
      float zg = zscr[(2 * 32 + jj_e) * 8 + b_e] + bi2;
      float zo = zscr[(3 * 32 + jj_e) * 8 + b_e] + bi3;
      float cp = cstate[jj_e * 8 + b_e];
      float cn = sigmoidf_(zf) * cp + sigmoidf_(zi) * tanhf(zg);
      float hn = sigmoidf_(zo) * tanhf(cn);
      cstate[jj_e * 8 + b_e] = cn;
      int j = j0 + jj_e;
      hN[b_e * 512 + j] = (bf16_t)hn;                       // next-step A input
      hs[((size_t)t * 8 + b_e) * 512 + j] = hn;             // full sequence out
      if (s == S_LEN - 1) {
        out[OUT_HN + dir * 4096 + b_e * 512 + j] = hn;
        out[OUT_CN + dir * 4096 + b_e * 512 + j] = cn;
      }
    }

    // cross-WG barrier over this direction's 16 WGs (monotonic, no reset)
    __syncthreads();
    if (tid == 0) {
      __threadfence();  // release h writes to device scope
      unsigned prev = atomicAdd(cntp, 1u);
      unsigned target = (unsigned)NWG_PER_DIR * (unsigned)(s + 1);
      if (prev + 1u == target) {
        __hip_atomic_store(genp, target, __ATOMIC_RELEASE,
                           __HIP_MEMORY_SCOPE_AGENT);
      } else {
        while (__hip_atomic_load(genp, __ATOMIC_ACQUIRE,
                                 __HIP_MEMORY_SCOPE_AGENT) < target) {
          __builtin_amdgcn_s_sleep(2);
        }
      }
      __threadfence();  // acquire: invalidate stale lines before next reads
    }
    __syncthreads();
  }
}

// ---------------------------------------------------------------------------
__global__ void __launch_bounds__(256)
combine_kernel(const float* __restrict__ hsf, const float* __restrict__ hsb,
               const int* __restrict__ comb_idx, const int* __restrict__ comb_mask,
               const long long* __restrict__ order, float* __restrict__ out)
{
  int bs = blockIdx.x;                  // over (b, s') = 8 x 1022
  int b  = bs / 1022;
  int sp = bs % 1022;
  int s  = sp + 1;
  int ob = (int)order[b];               // reference: idx row & out row = order[b]

  __shared__ int   sidx[8];
  __shared__ float smsk[8];
  if (threadIdx.x < 8) {
    int k = threadIdx.x;
    size_t base = ((size_t)ob * 1024 + s) * 8 + k;
    sidx[k] = comb_idx[base];
    smsk[k] = (float)comb_mask[base];
  }
  __syncthreads();

  float cntv = 0.f;
#pragma unroll
  for (int k = 0; k < 8; ++k) cntv += smsk[k];
  float scale = 1.0f / (1.0f + cntv);

  int c0 = threadIdx.x * 4;             // 4 feature cols per thread (of 1024)
  const float* src = (c0 < 512) ? hsf : hsb;
  int cc = (c0 < 512) ? c0 : (c0 - 512);

  float4 acc = *(const float4*)(src + ((size_t)s * 8 + b) * 512 + cc);
#pragma unroll
  for (int k = 0; k < 8; ++k) {
    if (smsk[k] != 0.f) {
      float4 g = *(const float4*)(src + ((size_t)sidx[k] * 8 + b) * 512 + cc);
      acc.x += g.x; acc.y += g.y; acc.z += g.z; acc.w += g.w;
    }
  }
  float4 r;
  r.x = acc.x * scale; r.y = acc.y * scale; r.z = acc.z * scale; r.w = acc.w * scale;
  *(float4*)(out + ((size_t)ob * 1022 + sp) * 1024 + c0) = r;
}

// ---------------------------------------------------------------------------
extern "C" void kernel_launch(void* const* d_in, const int* in_sizes, int n_in,
                              void* d_out, int out_size, void* d_ws, size_t ws_size,
                              hipStream_t stream) {
  const float*     input_t   = (const float*)d_in[0];
  const int*       comb_idx  = (const int*)d_in[1];
  const int*       comb_mask = (const int*)d_in[2];
  const long long* order     = (const long long*)d_in[3];  // int64
  const float*     w_ih_f    = (const float*)d_in[4];
  const float*     w_hh_f    = (const float*)d_in[5];
  const float*     b_f       = (const float*)d_in[6];
  const float*     w_ih_b    = (const float*)d_in[7];
  const float*     w_hh_b    = (const float*)d_in[8];
  const float*     b_b       = (const float*)d_in[9];
  float* out = (float*)d_out;
  char*  ws  = (char*)d_ws;

  bf16_t*   xbf  = (bf16_t*)(ws + X_BF_OFF);
  bf16_t*   hbuf = (bf16_t*)(ws + HBUF_OFF);
  float*    hsf  = (float*)(ws + HSF_OFF);
  float*    hsb  = (float*)(ws + HSB_OFF);
  unsigned* bar  = (unsigned*)(ws + BAR_OFF);

  init_small<<<1, 256, 0, stream>>>(hbuf, bar);
  build_xbf<<<(S_LEN * 16 * D_SZ) / 256, 256, 0, stream>>>(input_t, xbf);

  size_t smem = LDS_WIH + LDS_ZSCR + LDS_CST;  // 136,192 B
  lstm_kernel<<<2 * NWG_PER_DIR, THREADS_LSTM, smem, stream>>>(
      xbf, hbuf, hsf, hsb,
      w_ih_f, w_hh_f, b_f, w_ih_b, w_hh_b, b_b,
      bar, out);

  combine_kernel<<<8 * 1022, 256, 0, stream>>>(hsf, hsb, comb_idx, comb_mask,
                                               order, out);
}
